// TransferBlock_21139829031078
// MI455X (gfx1250) — matrix operand order repaired and checked by reading, not compiled, not run on hardware
//
#include <hip/hip_runtime.h>
#include <hip/hip_bf16.h>

// Problem constants (reference: B=2, T=2048, E=768, H=12, D=64)
#define B_  2
#define T_  2048
#define E_  768
#define H_  12
#define D_  64
#define M_  (B_*T_)     // 4096 rows
#define FF_ (4*E_)      // 3072

typedef __attribute__((ext_vector_type(16))) __bf16 v16bf;
typedef __attribute__((ext_vector_type(8)))  float  v8f;

union Frag16 { uint4 q[2]; unsigned short s[16]; v16bf v; };
union Acc8   { v8f v; float f[8]; };

__device__ __forceinline__ unsigned short f2bf(float f) {
    unsigned int u = __float_as_uint(f);
    unsigned int r = u + 0x7FFFu + ((u >> 16) & 1u);   // round-to-nearest-even
    return (unsigned short)(r >> 16);
}

__device__ __forceinline__ float gelu_f(float x) {
    const float c = 0.7978845608028654f;               // sqrt(2/pi)
    float x3 = x * x * x;
    return 0.5f * x * (1.0f + tanhf(c * (x + 0.044715f * x3)));
}

__device__ __forceinline__ v8f wmma_bf16(v16bf a, v16bf b, v8f c) {
    return __builtin_amdgcn_wmma_f32_16x16x32_bf16(
        /*neg_a=*/false, a, /*neg_b=*/false, b,
        /*c_mod=*/(short)0, c, /*reuse_a=*/false, /*reuse_b=*/false);
}

// CDNA5 async global->LDS copy (no VGPR staging, tracked with ASYNCcnt).
// Generic LDS pointer's low 32 bits are the LDS byte address (ISA aperture rule).
__device__ __forceinline__ void async_load_b128(const unsigned short* gptr,
                                                unsigned short* lptr) {
    unsigned int lds = (unsigned int)(unsigned long long)lptr;
    asm volatile("global_load_async_to_lds_b128 %0, %1, off"
                 :: "v"(lds), "v"(gptr) : "memory");
}
__device__ __forceinline__ void wait_async_le8() {
    asm volatile("s_wait_asynccnt 0x8" ::: "memory");
}
__device__ __forceinline__ void wait_async_0() {
    asm volatile("s_wait_asynccnt 0x0" ::: "memory");
}

// ---------------------------------------------------------------------------
// Weight convert + transpose:  W[K][N] fp32  ->  Wt[N][K] bf16
// ---------------------------------------------------------------------------
__global__ void __launch_bounds__(256)
k_conv_transpose(const float* __restrict__ W, unsigned short* __restrict__ Wt,
                 int K, int N) {
    int i = blockIdx.x * 256 + threadIdx.x;   // grid sized exactly K*N/256
    int k = i / N;
    int n = i - k * N;
    Wt[(size_t)n * K + k] = f2bf(W[i]);
}

// ---------------------------------------------------------------------------
// LayerNorm (ddof=1 variance) -> bf16 output; one 256-thread block per row
// ---------------------------------------------------------------------------
__global__ void __launch_bounds__(256)
k_layernorm(const float* __restrict__ x, const float* __restrict__ sc,
            const float* __restrict__ sh, unsigned short* __restrict__ out) {
    int row = blockIdx.x;
    const float* xr = x + (size_t)row * E_;
    float v[3], s = 0.f, sq = 0.f;
#pragma unroll
    for (int j = 0; j < 3; j++) {
        v[j] = xr[threadIdx.x + j * 256];
        s += v[j]; sq += v[j] * v[j];
    }
#pragma unroll
    for (int off = 1; off < 32; off <<= 1) {
        s  += __shfl_xor(s,  off, 32);
        sq += __shfl_xor(sq, off, 32);
    }
    __shared__ float rs[8], rq[8];
    int wv = threadIdx.x >> 5, ln = threadIdx.x & 31;
    if (ln == 0) { rs[wv] = s; rq[wv] = sq; }
    __syncthreads();
    s = 0.f; sq = 0.f;
#pragma unroll
    for (int i = 0; i < 8; i++) { s += rs[i]; sq += rq[i]; }
    float mean = s * (1.0f / E_);
    float var  = (sq - s * mean) * (1.0f / (E_ - 1));   // unbiased (ddof=1)
    float inv  = rsqrtf(var + 1e-5f);
#pragma unroll
    for (int j = 0; j < 3; j++) {
        int e = threadIdx.x + j * 256;
        out[(size_t)row * E_ + e] = f2bf(sc[e] * (v[j] - mean) * inv + sh[e]);
    }
}

// ---------------------------------------------------------------------------
// bf16 WMMA GEMM:  C[M,N] = A[M,K](bf16) * Wt[N,K](bf16) + bias
//   mode 0: fp32 out = acc + bias + resid
//   mode 1: bf16 out = gelu(acc + bias)
//   mode 2: bf16 out = acc
// Block tile 128x128, 8 waves (4M x 2N), wave tile 32x64 (8 WMMA accs).
// K-step 64 per iteration, double-buffered LDS filled with
// global_load_async_to_lds_b128 (ASYNCcnt), prefetching the next tile.
// ---------------------------------------------------------------------------
__global__ void __launch_bounds__(256)
k_gemm(const unsigned short* __restrict__ A, const unsigned short* __restrict__ Bt,
       const float* __restrict__ bias, const float* __restrict__ resid,
       float* __restrict__ outF, unsigned short* __restrict__ outH,
       int M, int N, int K, int mode) {
    __shared__ __align__(16) unsigned short lA[2][128 * 64];
    __shared__ __align__(16) unsigned short lB[2][128 * 64];

    int tid = threadIdx.x;
    int lane = tid & 31, wv = tid >> 5;
    int wm = wv & 3;            // 4 M-wave groups of 32 rows
    int wn = wv >> 2;           // 2 N-wave groups of 64 cols
    int bm = blockIdx.y * 128, bn = blockIdx.x * 128;

    Acc8 acc[2][4];
#pragma unroll
    for (int mg = 0; mg < 2; mg++)
#pragma unroll
        for (int ng = 0; ng < 4; ng++)
#pragma unroll
            for (int i = 0; i < 8; i++) acc[mg][ng].f[i] = 0.f;

    // copy coords: each thread moves 4x b128 per tile per matrix
    int crow = tid >> 3;              // 0..31
    int ccol = (tid & 7) * 8;         // 0..56
    const unsigned short* gA = A  + (size_t)(bm + crow) * K + ccol;
    const unsigned short* gB = Bt + (size_t)(bn + crow) * K + ccol;

    int klo   = (lane & 16) ? 8 : 0;   // A-frag K interleave
    int klb   = (lane & 16) ? 8 : 0;   // B-frag K interleave
    int mhalf = (lane & 16) ? 8 : 0;   // C-layout row half
    int l15   = lane & 15;

    // prologue: async-fill buffer 0
#pragma unroll
    for (int c = 0; c < 4; c++) {
        async_load_b128(gA + (size_t)(c * 32) * K, &lA[0][(c * 32 + crow) * 64 + ccol]);
        async_load_b128(gB + (size_t)(c * 32) * K, &lB[0][(c * 32 + crow) * 64 + ccol]);
    }

    int nk = K >> 6;                  // K / 64
    for (int kt = 0; kt < nk; kt++) {
        int cur = kt & 1;
        bool pre = (kt + 1) < nk;
        if (pre) {                    // prefetch next tile into other buffer
            const unsigned short* pA = gA + (size_t)(kt + 1) * 64;
            const unsigned short* pB = gB + (size_t)(kt + 1) * 64;
#pragma unroll
            for (int c = 0; c < 4; c++) {
                async_load_b128(pA + (size_t)(c * 32) * K, &lA[1 - cur][(c * 32 + crow) * 64 + ccol]);
                async_load_b128(pB + (size_t)(c * 32) * K, &lB[1 - cur][(c * 32 + crow) * 64 + ccol]);
            }
            wait_async_le8();         // current tile's 8 done (in-order), next 8 in flight
        } else {
            wait_async_0();
        }
        __syncthreads();              // all waves' current-tile data landed

        const unsigned short* pa = &lA[cur][0];
        const unsigned short* pb = &lB[cur][0];
#pragma unroll
        for (int ks = 0; ks < 2; ks++) {
            Frag16 fa0, fa1;
            int r0 = (wm * 32 + l15) * 64 + ks * 32;
            fa0.q[0] = *(const uint4*)&pa[r0 + klo];
            fa0.q[1] = *(const uint4*)&pa[r0 + klo + 16];
            fa1.q[0] = *(const uint4*)&pa[r0 + 16 * 64 + klo];
            fa1.q[1] = *(const uint4*)&pa[r0 + 16 * 64 + klo + 16];
#pragma unroll
            for (int ng = 0; ng < 4; ng++) {
                int nrow = (wn * 64 + ng * 16 + l15) * 64 + ks * 32;
                Frag16 fb;
                fb.q[0] = *(const uint4*)&pb[nrow + klb];
                fb.q[1] = *(const uint4*)&pb[nrow + klb + 16];
                acc[0][ng].v = wmma_bf16(fa0.v, fb.v, acc[0][ng].v);
                acc[1][ng].v = wmma_bf16(fa1.v, fb.v, acc[1][ng].v);
            }
        }
        __syncthreads();              // reads done before buffer is refilled
    }

    // epilogue
    int mbase = bm + wm * 32 + mhalf;
#pragma unroll
    for (int ng = 0; ng < 4; ng++) {
        int nc = bn + wn * 64 + ng * 16 + l15;
        float bv = bias ? bias[nc] : 0.f;
#pragma unroll
        for (int mg = 0; mg < 2; mg++) {
#pragma unroll
            for (int r = 0; r < 8; r++) {
                size_t idx = (size_t)(mbase + mg * 16 + r) * N + nc;
                float val = acc[mg][ng].f[r] + bv;
                if (mode == 0)      outF[idx] = val + (resid ? resid[idx] : 0.f);
                else if (mode == 1) outH[idx] = f2bf(gelu_f(val));
                else                outH[idx] = f2bf(val);
            }
        }
    }
}

// ---------------------------------------------------------------------------
// Causal flash attention. Q/K/V are bf16 [B,T,E] (head slice = h*64).
// Block = 128 q-rows (8 waves x 16 rows), streams 64-key tiles.
// ---------------------------------------------------------------------------
__global__ void __launch_bounds__(256)
k_attn(const unsigned short* __restrict__ Q, const unsigned short* __restrict__ Kk,
       const unsigned short* __restrict__ V, unsigned short* __restrict__ ctx) {
    __shared__ __align__(16) unsigned short lK[64 * 64];    // [key][d]
    __shared__ __align__(16) unsigned short lVt[64 * 64];   // [d][key]
    __shared__ __align__(16) unsigned short lP[8][16 * 64]; // per-wave P relayout

    int tid = threadIdx.x, lane = tid & 31, wv = tid >> 5;
    int b = blockIdx.y / H_, h = blockIdx.y % H_;
    int q0 = blockIdx.x * 128;
    int qrow = q0 + wv * 16;
    size_t headoff = (size_t)h * D_;

    int m     = lane & 15;
    int klo   = (lane & 16) ? 8 : 0;
    int klb   = (lane & 16) ? 8 : 0;
    int mhalf = (lane & 16) ? 8 : 0;

    const unsigned short* qp = Q + ((size_t)(b * T_ + qrow + m)) * E_ + headoff;
    Frag16 fq0, fq1;
    fq0.q[0] = *(const uint4*)&qp[klo];
    fq0.q[1] = *(const uint4*)&qp[klo + 16];
    fq1.q[0] = *(const uint4*)&qp[32 + klo];
    fq1.q[1] = *(const uint4*)&qp[32 + klo + 16];

    Acc8 o[4];
    float rowm[8], rowl[8];
#pragma unroll
    for (int r = 0; r < 8; r++) {
        rowm[r] = -3.0e38f; rowl[r] = 0.f;
#pragma unroll
        for (int f = 0; f < 4; f++) o[f].f[r] = 0.f;
    }

    int ntiles = q0 / 64 + 2;
    for (int kt = 0; kt < ntiles; kt++) {
        int k0 = kt * 64;
        {   // cooperative tile load: 64 keys x 64 d; V stored transposed
            int key = tid >> 2;
            int d0  = (tid & 3) * 16;
            const unsigned short* kp = Kk + ((size_t)(b * T_ + k0 + key)) * E_ + headoff + d0;
            *(uint4*)&lK[key * 64 + d0]     = *(const uint4*)&kp[0];
            *(uint4*)&lK[key * 64 + d0 + 8] = *(const uint4*)&kp[8];
            const unsigned short* vp = V + ((size_t)(b * T_ + k0 + key)) * E_ + headoff + d0;
            unsigned short tmp[16];
            *(uint4*)&tmp[0] = *(const uint4*)&vp[0];
            *(uint4*)&tmp[8] = *(const uint4*)&vp[8];
#pragma unroll
            for (int j = 0; j < 16; j++) lVt[(d0 + j) * 64 + key] = tmp[j];
        }
        __syncthreads();

        if (k0 <= qrow + 15) {                 // skip fully-masked tiles
            float sarr[4][8];
#pragma unroll
            for (int ng = 0; ng < 4; ng++) {   // S = Q @ K^T (16 x 64)
                int kn = ng * 16 + (lane & 15);
                Frag16 fb;
                Acc8 s;
#pragma unroll
                for (int i = 0; i < 8; i++) s.f[i] = 0.f;
                fb.q[0] = *(const uint4*)&lK[kn * 64 + klb];
                fb.q[1] = *(const uint4*)&lK[kn * 64 + klb + 16];
                s.v = wmma_bf16(fq0.v, fb.v, s.v);
                fb.q[0] = *(const uint4*)&lK[kn * 64 + 32 + klb];
                fb.q[1] = *(const uint4*)&lK[kn * 64 + 32 + klb + 16];
                s.v = wmma_bf16(fq1.v, fb.v, s.v);
#pragma unroll
                for (int r = 0; r < 8; r++) {
                    int qi = qrow + r + mhalf;
                    int ki = k0 + kn;
                    float sv = s.f[r] * 0.125f;             // 1/sqrt(64)
                    sarr[ng][r] = (ki > qi) ? -1e30f : sv;  // causal mask
                }
            }
            // online softmax (rows live in VGPR slots of the C layout)
#pragma unroll
            for (int r = 0; r < 8; r++) {
                float tm = sarr[0][r];
#pragma unroll
                for (int ng = 1; ng < 4; ng++) tm = fmaxf(tm, sarr[ng][r]);
#pragma unroll
                for (int off = 1; off < 16; off <<= 1)
                    tm = fmaxf(tm, __shfl_xor(tm, off, 32));
                float nm = fmaxf(rowm[r], tm);
                float corr = __expf(rowm[r] - nm);
                rowm[r] = nm;
                float ts = 0.f;
#pragma unroll
                for (int ng = 0; ng < 4; ng++) {
                    float p = __expf(sarr[ng][r] - nm);
                    sarr[ng][r] = p; ts += p;
                }
#pragma unroll
                for (int off = 1; off < 16; off <<= 1)
                    ts += __shfl_xor(ts, off, 32);
                rowl[r] = rowl[r] * corr + ts;
#pragma unroll
                for (int f = 0; f < 4; f++) o[f].f[r] *= corr;
            }
            // relayout P: C-layout -> A-layout via wave-private LDS scratch
            unsigned short* pw = &lP[wv][0];
#pragma unroll
            for (int ng = 0; ng < 4; ng++)
#pragma unroll
                for (int r = 0; r < 8; r++)
                    pw[(r + mhalf) * 64 + ng * 16 + (lane & 15)] = f2bf(sarr[ng][r]);
            Frag16 fp0, fp1;
            const unsigned short* pr = pw + (size_t)m * 64;
            fp0.q[0] = *(const uint4*)&pr[klo];
            fp0.q[1] = *(const uint4*)&pr[klo + 16];
            fp1.q[0] = *(const uint4*)&pr[32 + klo];
            fp1.q[1] = *(const uint4*)&pr[32 + klo + 16];
            // O += P @ V
#pragma unroll
            for (int dg = 0; dg < 4; dg++) {
                int dr = dg * 16 + (lane & 15);
                Frag16 fb;
                fb.q[0] = *(const uint4*)&lVt[dr * 64 + klb];
                fb.q[1] = *(const uint4*)&lVt[dr * 64 + klb + 16];
                o[dg].v = wmma_bf16(fp0.v, fb.v, o[dg].v);
                fb.q[0] = *(const uint4*)&lVt[dr * 64 + 32 + klb];
                fb.q[1] = *(const uint4*)&lVt[dr * 64 + 32 + klb + 16];
                o[dg].v = wmma_bf16(fp1.v, fb.v, o[dg].v);
            }
        }
        __syncthreads();
    }

#pragma unroll
    for (int dg = 0; dg < 4; dg++) {
        int d = dg * 16 + (lane & 15);
#pragma unroll
        for (int r = 0; r < 8; r++) {
            int t = qrow + r + mhalf;
            ctx[((size_t)(b * T_ + t)) * E_ + headoff + d] = f2bf(o[dg].f[r] / rowl[r]);
        }
    }
}

// ---------------------------------------------------------------------------
extern "C" void kernel_launch(void* const* d_in, const int* in_sizes, int n_in,
                              void* d_out, int out_size, void* d_ws, size_t ws_size,
                              hipStream_t stream) {
    const float* x    = (const float*)d_in[0];
    const float* wq   = (const float*)d_in[1];
    const float* wk   = (const float*)d_in[2];
    const float* wv   = (const float*)d_in[3];
    const float* wo   = (const float*)d_in[4];
    const float* bo   = (const float*)d_in[5];
    const float* ln1s = (const float*)d_in[6];
    const float* ln1b = (const float*)d_in[7];
    const float* ln2s = (const float*)d_in[8];
    const float* ln2b = (const float*)d_in[9];
    const float* w1   = (const float*)d_in[10];
    const float* b1   = (const float*)d_in[11];
    const float* w2   = (const float*)d_in[12];
    const float* b2   = (const float*)d_in[13];

    char* ws = (char*)d_ws;
    size_t off = 0;
    auto alloc = [&](size_t bytes) -> char* {
        char* p = ws + off;
        off += (bytes + 255) & ~(size_t)255;
        return p;
    };
    unsigned short* wqT = (unsigned short*)alloc((size_t)E_ * E_ * 2);
    unsigned short* wkT = (unsigned short*)alloc((size_t)E_ * E_ * 2);
    unsigned short* wvT = (unsigned short*)alloc((size_t)E_ * E_ * 2);
    unsigned short* woT = (unsigned short*)alloc((size_t)E_ * E_ * 2);
    unsigned short* w1T = (unsigned short*)alloc((size_t)E_ * FF_ * 2);
    unsigned short* w2T = (unsigned short*)alloc((size_t)E_ * FF_ * 2);
    unsigned short* h1  = (unsigned short*)alloc((size_t)M_ * E_ * 2);
    unsigned short* qb  = (unsigned short*)alloc((size_t)M_ * E_ * 2);
    unsigned short* kb  = (unsigned short*)alloc((size_t)M_ * E_ * 2);
    unsigned short* vb  = (unsigned short*)alloc((size_t)M_ * E_ * 2);
    unsigned short* cb  = (unsigned short*)alloc((size_t)M_ * E_ * 2);
    float*          x2  = (float*)         alloc((size_t)M_ * E_ * 4);
    unsigned short* h2  = (unsigned short*)alloc((size_t)M_ * E_ * 2);
    unsigned short* m1  = (unsigned short*)alloc((size_t)M_ * FF_ * 2);

    // 1) weight convert + transpose (bf16, [N][K])
    k_conv_transpose<<<E_ * E_ / 256, 256, 0, stream>>>(wq, wqT, E_, E_);
    k_conv_transpose<<<E_ * E_ / 256, 256, 0, stream>>>(wk, wkT, E_, E_);
    k_conv_transpose<<<E_ * E_ / 256, 256, 0, stream>>>(wv, wvT, E_, E_);
    k_conv_transpose<<<E_ * E_ / 256, 256, 0, stream>>>(wo, woT, E_, E_);
    k_conv_transpose<<<E_ * FF_ / 256, 256, 0, stream>>>(w1, w1T, E_, FF_);
    k_conv_transpose<<<E_ * FF_ / 256, 256, 0, stream>>>(w2, w2T, FF_, E_);

    dim3 blk(256);
    dim3 gE(E_ / 128, M_ / 128);    // (6, 32)
    dim3 gF(FF_ / 128, M_ / 128);   // (24, 32)

    // 2) LN1 -> h1 (bf16)
    k_layernorm<<<M_, blk, 0, stream>>>(x, ln1s, ln1b, h1);
    // 3) QKV projections
    k_gemm<<<gE, blk, 0, stream>>>(h1, wqT, nullptr, nullptr, nullptr, qb, M_, E_, E_, 2);
    k_gemm<<<gE, blk, 0, stream>>>(h1, wkT, nullptr, nullptr, nullptr, kb, M_, E_, E_, 2);
    k_gemm<<<gE, blk, 0, stream>>>(h1, wvT, nullptr, nullptr, nullptr, vb, M_, E_, E_, 2);
    // 4) causal flash attention
    k_attn<<<dim3(T_ / 128, B_ * H_), blk, 0, stream>>>(qb, kb, vb, cb);
    // 5) output projection + bias + residual -> x2 (fp32)
    k_gemm<<<gE, blk, 0, stream>>>(cb, woT, bo, x, x2, nullptr, M_, E_, E_, 0);
    // 6) LN2 -> h2 (bf16)
    k_layernorm<<<M_, blk, 0, stream>>>(x2, ln2s, ln2b, h2);
    // 7) MLP up + GELU -> m1 (bf16)
    k_gemm<<<gF, blk, 0, stream>>>(h2, w1T, b1, nullptr, nullptr, m1, M_, FF_, E_, 1);
    // 8) MLP down + bias + residual -> d_out (fp32)
    k_gemm<<<gE, blk, 0, stream>>>(m1, w2T, b2, x2, (float*)d_out, nullptr, M_, E_, FF_, 0);
}
